// NeurosynapticHybridAttention_86535001080335
// MI455X (gfx1250) — matrix operand order repaired and checked by reading, not compiled, not run on hardware
//
#include <hip/hip_runtime.h>
#include <hip/hip_bf16.h>
#include <math.h>

// ---------------- problem constants ----------------
constexpr int B  = 2;
constexpr int S  = 2048;
constexpr int D  = 1024;
constexpr int H  = 16;
constexpr int HD = 64;          // head dim
constexpr int M  = 16;          // persistent memory tokens
constexpr int T  = S + M;       // 2064
constexpr int Tp = 2080;        // padded to multiple of 32
constexpr float SCALE = 0.125f; // 64^-0.5
constexpr float NEGI  = -1.0e9f;
constexpr int HALF_WIN = 128;   // W // 2

// ---------------- WMMA helpers ----------------
typedef __attribute__((ext_vector_type(16))) _Float16 v16h;
typedef __attribute__((ext_vector_type(8)))  _Float16 v8h;
typedef __attribute__((ext_vector_type(8)))  float    v8f;

__device__ __forceinline__ v16h frag16(const _Float16* p0, const _Float16* p1) {
    v8h a = *(const v8h*)p0;
    v8h b = *(const v8h*)p1;
    v16h r;
#pragma unroll
    for (int i = 0; i < 8; ++i) { r[i] = a[i]; r[i + 8] = b[i]; }
    return r;
}

__device__ __forceinline__ v8f wmma_f16(v16h a, v16h b, v8f c) {
    return __builtin_amdgcn_wmma_f32_16x16x32_f16(
        /*neg_a=*/false, a, /*neg_b=*/false, b,
        /*c_mod=*/(short)0, c, /*reuse_a=*/false, /*reuse_b=*/false);
}

// max-reduction across the 16-lane half-wave row using DPP row rotates
// (row_ror:N = 0x120 + N); pure VALU, no LDS round trips.
template <int CTRL>
__device__ __forceinline__ float max_dpp_step(float x) {
    int t = __builtin_amdgcn_update_dpp(0, __float_as_int(x), CTRL, 0xf, 0xf, true);
    return fmaxf(x, __int_as_float(t));
}
__device__ __forceinline__ float half_max(float x) {
    x = max_dpp_step<0x121>(x);  // row_ror:1
    x = max_dpp_step<0x122>(x);  // row_ror:2
    x = max_dpp_step<0x124>(x);  // row_ror:4
    x = max_dpp_step<0x128>(x);  // row_ror:8
    return x;
}

// ---------------- QKV projection: Y = concat(mem, X) @ W + b ----------------
// Tile: 128 rows x 64 cols per block (256 threads = 8 waves; 16 rows/wave).
// K staged 64 at a time (2 WMMA k-steps per barrier round).
// MODE 0: Q -> (B,H,Tp,HD) f16, scaled by SCALE
// MODE 1: K -> (B,H,Tp,HD) f16
// MODE 2: V -> (B,H,HD,Tp) f16 (d-major so attention PV B-frags are contiguous)
constexpr int XSP = 72;  // LDS row stride (halves) for X tile  (128 x 64)
constexpr int WSP = 72;  // LDS row stride (halves) for W^T tile (64 x 64)

template <int MODE>
__global__ __launch_bounds__(256) void proj_kernel(
    const float* __restrict__ x,    // (B,S,D)
    const float* __restrict__ mem,  // (M,D)
    const float* __restrict__ Wm,   // (D,D)
    const float* __restrict__ bias, // (D)
    _Float16* __restrict__ outp) {
    __shared__ __align__(16) _Float16 Xs[128 * XSP];
    __shared__ __align__(16) _Float16 WsT[64 * WSP];

    const int tid  = threadIdx.x;
    const int wave = tid >> 5;
    const int lane = tid & 31;
    const int lq   = lane & 15;
    const int g    = lane >> 4;
    const int bx = blockIdx.x, by = blockIdx.y, bz = blockIdx.z; // by == head

    v8f z = {};
    v8f acc[4] = {z, z, z, z};

    for (int kk = 0; kk < D; kk += 64) {
        // stage X (concat(mem, x)) as f16: 32 halves / thread
        {
            int r  = tid >> 1;
            int c0 = (tid & 1) * 32;
            int t  = bx * 128 + r;
            float4 f[8];
            if (t < T) {
                const float* src = (t < M)
                    ? (mem + (size_t)t * D + kk + c0)
                    : (x + ((size_t)bz * S + (t - M)) * D + kk + c0);
                const float4* s4 = (const float4*)src;
#pragma unroll
                for (int q = 0; q < 8; ++q) f[q] = s4[q];
            } else {
#pragma unroll
                for (int q = 0; q < 8; ++q) f[q] = make_float4(0.f, 0.f, 0.f, 0.f);
            }
            _Float16* dst = Xs + r * XSP + c0;
#pragma unroll
            for (int q = 0; q < 8; ++q) {
                dst[q * 4 + 0] = (_Float16)f[q].x;
                dst[q * 4 + 1] = (_Float16)f[q].y;
                dst[q * 4 + 2] = (_Float16)f[q].z;
                dst[q * 4 + 3] = (_Float16)f[q].w;
            }
        }
        // stage W transposed: WsT[col][k], 16 values / thread
        {
            int r  = tid >> 2;          // k row within tile, 0..63
            int c0 = (tid & 3) * 16;    // col within 64
            const float* src = Wm + (size_t)(kk + r) * D + by * 64 + c0;
            const float4* s4 = (const float4*)src;
#pragma unroll
            for (int q = 0; q < 4; ++q) {
                float4 gq = s4[q];
                WsT[(c0 + q * 4 + 0) * WSP + r] = (_Float16)gq.x;
                WsT[(c0 + q * 4 + 1) * WSP + r] = (_Float16)gq.y;
                WsT[(c0 + q * 4 + 2) * WSP + r] = (_Float16)gq.z;
                WsT[(c0 + q * 4 + 3) * WSP + r] = (_Float16)gq.w;
            }
        }
        __syncthreads();

        const _Float16* ar = Xs + (wave * 16 + lq) * XSP;
#pragma unroll
        for (int ks = 0; ks < 64; ks += 32) {
            v16h a = frag16(ar + ks + 16 * g, ar + ks + 16 * g + 8);
#pragma unroll
            for (int nt = 0; nt < 4; ++nt) {
                const _Float16* br = WsT + (nt * 16 + lq) * WSP + ks + 16 * g;
                v16h bb = frag16(br, br + 8);
                acc[nt] = wmma_f16(a, bb, acc[nt]);
            }
        }
        __syncthreads();
    }

    // epilogue: bias, (scale), scatter to head layout
#pragma unroll
    for (int nt = 0; nt < 4; ++nt) {
        float bcol = bias[by * 64 + nt * 16 + lq];
#pragma unroll
        for (int r = 0; r < 8; ++r) {
            int t = bx * 128 + wave * 16 + r + 8 * g;
            if (t < T) {
                float v = acc[nt][r] + bcol;
                if (MODE == 0) v *= SCALE;
                if (MODE == 2) {
                    outp[((size_t)(bz * H + by) * HD + nt * 16 + lq) * Tp + t] = (_Float16)v;
                } else {
                    outp[((size_t)(bz * H + by) * Tp + t) * HD + nt * 16 + lq] = (_Float16)v;
                }
            }
        }
    }
}

// ---------------- banded flash attention ----------------
// grid (S/128, H, B); 8 waves/block; each wave owns 16 query rows.
// Softmax denominator comes "for free" from a 5th WMMA accumulator
// (P x ones-column), so no cross-lane sum reductions are needed.
__global__ __launch_bounds__(256) void attn_kernel(
    const _Float16* __restrict__ Qh,  // (B,H,Tp,HD), pre-scaled
    const _Float16* __restrict__ Kh,  // (B,H,Tp,HD)
    const _Float16* __restrict__ Vt,  // (B,H,HD,Tp)
    _Float16* __restrict__ ctx) {     // (B,S,D)
    __shared__ __align__(16) _Float16 Pbuf[8][16 * 32];

    const int tid  = threadIdx.x;
    const int wave = tid >> 5;
    const int lane = tid & 31;
    const int lq   = lane & 15;
    const int g    = lane >> 4;
    const int bx = blockIdx.x, by = blockIdx.y, bz = blockIdx.z;

    const size_t bh = (size_t)bz * H + by;
    const _Float16* Qb = Qh + bh * Tp * HD;
    const _Float16* Kb = Kh + bh * Tp * HD;
    const _Float16* Vb = Vt + bh * HD * Tp;

    const int i0 = M + bx * 128 + wave * 16;   // global query row (incl. mem offset)

    // Q A-fragments (d = 0..31 and 32..63)
    const _Float16* qrow = Qb + (size_t)(i0 + lq) * HD;
    v16h qa0 = frag16(qrow + 16 * g,      qrow + 16 * g + 8);
    v16h qa1 = frag16(qrow + 32 + 16 * g, qrow + 40 + 16 * g);

    // constant-1 B fragment: makes o4 accumulate rowsum(P)
    v16h vones;
#pragma unroll
    for (int i = 0; i < 16; ++i) vones[i] = (_Float16)1.0f;

    v8f z = {};
    v8f o0 = z, o1 = z, o2 = z, o3 = z, o4 = z;
    float mrow[8];
#pragma unroll
    for (int r = 0; r < 8; ++r) mrow[r] = -1.0e30f;

    // band of key tiles (32 keys each) + the mem tile at j0 = 0
    int lo = i0 - HALF_WIN;          lo = (lo < 32) ? 32 : (lo & ~31);
    int hi = i0 + 15 + HALF_WIN + 1; hi = (hi + 31) & ~31; if (hi > Tp) hi = Tp;
    const int ntiles = (hi - lo) / 32 + 1;

    for (int tt = 0; tt < ntiles; ++tt) {
        const int j0 = (tt == 0) ? 0 : (lo + (tt - 1) * 32);

        // K B-fragments: keys j0..j0+15 / j0+16..j0+31, d-chunks 0..31 / 32..63
        const _Float16* k0r = Kb + (size_t)(j0 + lq) * HD;
        const _Float16* k1r = Kb + (size_t)(j0 + 16 + lq) * HD;
        v16h kb00 = frag16(k0r + 16 * g,      k0r + 16 * g + 8);
        v16h kb01 = frag16(k0r + 32 + 16 * g, k0r + 40 + 16 * g);
        v16h kb10 = frag16(k1r + 16 * g,      k1r + 16 * g + 8);
        v16h kb11 = frag16(k1r + 32 + 16 * g, k1r + 40 + 16 * g);

        v8f s0 = z, s1 = z;
        s0 = wmma_f16(qa0, kb00, s0);
        s0 = wmma_f16(qa1, kb01, s0);
        s1 = wmma_f16(qa0, kb10, s1);
        s1 = wmma_f16(qa1, kb11, s1);

        // tile fully inside the band (and fully valid)? -> skip all masking ALU
        const bool full = (j0 >= i0 - 113) && (j0 <= i0 + 97) && (j0 + 31 < T);

        float nm[8];
        if (full) {
#pragma unroll
            for (int r = 0; r < 8; ++r) {
                float mx = half_max(fmaxf(s0[r], s1[r]));
                nm[r] = fmaxf(mrow[r], mx);
            }
        } else {
            const int j0c = j0 + lq;
            const int j1c = j0 + 16 + lq;
#pragma unroll
            for (int r = 0; r < 8; ++r) {
                int i = i0 + r + 8 * g;
                int d0 = i - j0c, d1 = i - j1c;
                bool a0 = (j0c < M) || (j0c < T && d0 <= HALF_WIN && d0 >= -HALF_WIN);
                bool a1 = (j1c < M) || (j1c < T && d1 <= HALF_WIN && d1 >= -HALF_WIN);
                float x0 = a0 ? s0[r] : NEGI;
                float x1 = a1 ? s1[r] : NEGI;
                s0[r] = x0; s1[r] = x1;
                float mx = half_max(fmaxf(x0, x1));
                nm[r] = fmaxf(mrow[r], mx);
            }
        }

#pragma unroll
        for (int r = 0; r < 8; ++r) {
            float alpha = __expf(mrow[r] - nm[r]);
            float p0 = __expf(s0[r] - nm[r]);
            float p1 = __expf(s1[r] - nm[r]);
            mrow[r] = nm[r];
            o0[r] *= alpha; o1[r] *= alpha; o2[r] *= alpha;
            o3[r] *= alpha; o4[r] *= alpha;
            int prow = r + 8 * g;
            Pbuf[wave][prow * 32 + lq]      = (_Float16)p0;
            Pbuf[wave][prow * 32 + 16 + lq] = (_Float16)p1;
        }
        asm volatile("s_wait_dscnt 0" ::: "memory");

        // P (16x32) as A-fragment; V B-fragments from d-major Vt
        const _Float16* pr = &Pbuf[wave][lq * 32];
        v16h pa = frag16(pr + 16 * g, pr + 16 * g + 8);

        const _Float16* v0 = Vb + (size_t)(0  + lq) * Tp + j0 + 16 * g;
        const _Float16* v1 = Vb + (size_t)(16 + lq) * Tp + j0 + 16 * g;
        const _Float16* v2 = Vb + (size_t)(32 + lq) * Tp + j0 + 16 * g;
        const _Float16* v3 = Vb + (size_t)(48 + lq) * Tp + j0 + 16 * g;
        o0 = wmma_f16(pa, frag16(v0, v0 + 8), o0);
        o1 = wmma_f16(pa, frag16(v1, v1 + 8), o1);
        o2 = wmma_f16(pa, frag16(v2, v2 + 8), o2);
        o3 = wmma_f16(pa, frag16(v3, v3 + 8), o3);
        o4 = wmma_f16(pa, vones, o4);            // row sums of P
    }

    // normalize by the WMMA-accumulated row sum + write ctx (drop mem rows)
#pragma unroll
    for (int r = 0; r < 8; ++r) {
        float inv = 1.0f / o4[r];
        int srow = i0 + r + 8 * g - M;
        _Float16* crow = ctx + ((size_t)bz * S + srow) * D + by * HD;
        crow[0  + lq] = (_Float16)(o0[r] * inv);
        crow[16 + lq] = (_Float16)(o1[r] * inv);
        crow[32 + lq] = (_Float16)(o2[r] * inv);
        crow[48 + lq] = (_Float16)(o3[r] * inv);
    }
}

// ---------------- output projection: out = ctx @ Wo + bo (f32 out) ----------------
__global__ __launch_bounds__(256) void oproj_kernel(
    const _Float16* __restrict__ ctxp, // (B,S,D) f16
    const float* __restrict__ Wm,      // (D,D)
    const float* __restrict__ bias,    // (D)
    float* __restrict__ outp) {        // (B,S,D) f32
    __shared__ __align__(16) _Float16 Xs[128 * XSP];
    __shared__ __align__(16) _Float16 WsT[64 * WSP];

    const int tid  = threadIdx.x;
    const int wave = tid >> 5;
    const int lane = tid & 31;
    const int lq   = lane & 15;
    const int g    = lane >> 4;
    const int bx = blockIdx.x, by = blockIdx.y, bz = blockIdx.z;

    v8f z = {};
    v8f acc[4] = {z, z, z, z};

    for (int kk = 0; kk < D; kk += 64) {
        {
            int r  = tid >> 1;
            int c0 = (tid & 1) * 32;
            const _Float16* src = ctxp + ((size_t)bz * S + bx * 128 + r) * D + kk + c0;
            _Float16* dst = Xs + r * XSP + c0;
            *(v8h*)(dst)      = *(const v8h*)(src);
            *(v8h*)(dst + 8)  = *(const v8h*)(src + 8);
            *(v8h*)(dst + 16) = *(const v8h*)(src + 16);
            *(v8h*)(dst + 24) = *(const v8h*)(src + 24);
        }
        {
            int r  = tid >> 2;
            int c0 = (tid & 3) * 16;
            const float* src = Wm + (size_t)(kk + r) * D + by * 64 + c0;
            const float4* s4 = (const float4*)src;
#pragma unroll
            for (int q = 0; q < 4; ++q) {
                float4 gq = s4[q];
                WsT[(c0 + q * 4 + 0) * WSP + r] = (_Float16)gq.x;
                WsT[(c0 + q * 4 + 1) * WSP + r] = (_Float16)gq.y;
                WsT[(c0 + q * 4 + 2) * WSP + r] = (_Float16)gq.z;
                WsT[(c0 + q * 4 + 3) * WSP + r] = (_Float16)gq.w;
            }
        }
        __syncthreads();

        const _Float16* ar = Xs + (wave * 16 + lq) * XSP;
#pragma unroll
        for (int ks = 0; ks < 64; ks += 32) {
            v16h a = frag16(ar + ks + 16 * g, ar + ks + 16 * g + 8);
#pragma unroll
            for (int nt = 0; nt < 4; ++nt) {
                const _Float16* br = WsT + (nt * 16 + lq) * WSP + ks + 16 * g;
                v16h bb = frag16(br, br + 8);
                acc[nt] = wmma_f16(a, bb, acc[nt]);
            }
        }
        __syncthreads();
    }

#pragma unroll
    for (int nt = 0; nt < 4; ++nt) {
        float bcol = bias[by * 64 + nt * 16 + lq];
#pragma unroll
        for (int r = 0; r < 8; ++r) {
            int t = bx * 128 + wave * 16 + r + 8 * g;
            outp[((size_t)bz * S + t) * D + by * 64 + nt * 16 + lq] = acc[nt][r] + bcol;
        }
    }
}

// ---------------- host launch ----------------
extern "C" void kernel_launch(void* const* d_in, const int* in_sizes, int n_in,
                              void* d_out, int out_size, void* d_ws, size_t ws_size,
                              hipStream_t stream) {
    const float* query = (const float*)d_in[0];
    const float* key   = (const float*)d_in[1];
    const float* value = (const float*)d_in[2];
    const float* pmem  = (const float*)d_in[3];
    const float* Wq = (const float*)d_in[4];
    const float* bq = (const float*)d_in[5];
    const float* Wk = (const float*)d_in[6];
    const float* bk = (const float*)d_in[7];
    const float* Wv = (const float*)d_in[8];
    const float* bv = (const float*)d_in[9];
    const float* Wo = (const float*)d_in[10];
    const float* bo = (const float*)d_in[11];
    float* out = (float*)d_out;

    const size_t QKV = (size_t)B * H * Tp * HD;   // f16 elements per tensor
    _Float16* Qh  = (_Float16*)d_ws;
    _Float16* Kh  = Qh + QKV;
    _Float16* Vt  = Kh + QKV;
    _Float16* Ctx = Vt + QKV;

    // zero-fill so the Tp padding rows contribute exactly zero
    hipMemsetAsync(d_ws, 0, 3 * QKV * sizeof(_Float16), stream);

    dim3 blk(256);
    dim3 gproj((T + 127) / 128, D / 64, B);   // 17 x 16 x 2
    proj_kernel<0><<<gproj, blk, 0, stream>>>(query, pmem, Wq, bq, Qh);
    proj_kernel<1><<<gproj, blk, 0, stream>>>(key,   pmem, Wk, bk, Kh);
    proj_kernel<2><<<gproj, blk, 0, stream>>>(value, pmem, Wv, bv, Vt);

    dim3 gattn(S / 128, H, B);                // 16 x 16 x 2
    attn_kernel<<<gattn, blk, 0, stream>>>(Qh, Kh, Vt, Ctx);

    dim3 gout(S / 128, D / 64, B);            // 16 x 16 x 2
    oproj_kernel<<<gout, blk, 0, stream>>>(Ctx, Wo, bo, out);
}
